// EncoderDecoderLSTM_5068061409646
// MI455X (gfx1250) — compile-verified, matrix operand order checked by reading, not run on hardware
//
#include <hip/hip_runtime.h>
#include <hip/hip_bf16.h>

#define HID  256
#define TENC 336
#define TDEC 168
#define MATELEMS 262144   // 1024 * 256

typedef __attribute__((ext_vector_type(16))) __bf16 v16bf;
typedef __attribute__((ext_vector_type(8)))  float  v8f;

__device__ __forceinline__ unsigned short f2bf(float f) {
  unsigned int u = __float_as_uint(f);
  unsigned int r = u + 0x7FFFu + ((u >> 16) & 1u);   // round-to-nearest-even
  return (unsigned short)(r >> 16);
}
__device__ __forceinline__ float bf2f(unsigned short h) {
  return __uint_as_float(((unsigned int)h) << 16);
}
__device__ __forceinline__ float sigm(float x) { return 1.0f / (1.0f + __expf(-x)); }
__device__ __forceinline__ float tanh_fast(float x) { return 1.0f - 2.0f / (__expf(2.0f * x) + 1.0f); }

// ---------------------------------------------------------------------------
// Pack fp32 weight (4H x H, row-major; gates = h @ W^T) into bf16 WMMA-B tile
// layout: per (16-col N-tile, 32-deep K-chunk) a 1 KB block where lane l of a
// wave loads its 16 bf16 B-fragment values with one contiguous 32-byte read.
// lane = 16*(k-half) + (n%16); within lane: 16 consecutive K values.
// ---------------------------------------------------------------------------
__global__ __launch_bounds__(256)
void pack_weights(const float* __restrict__ src, unsigned short* __restrict__ dst) {
  int idx = blockIdx.x * blockDim.x + threadIdx.x;   // idx = n*256 + k
  int n = idx >> 8, k = idx & 255;
  int nt = n >> 4, nl = n & 15;
  int kc = k >> 5, kk = k & 31;
  int half = kk >> 4, j = kk & 15;
  int lane = (half << 4) | nl;
  dst[(((nt << 3) + kc) << 9) + (lane << 4) + j] = f2bf(src[idx]);
}

// B fragment: contiguous 32B per lane from packed weights (global / L2)
__device__ __forceinline__ v16bf loadB(const unsigned short* wmat, int lane, int nt, int kc) {
  return *(const v16bf*)(wmat + ((((nt << 3) + kc) << 9) + (lane << 4)));
}
// A fragment: h tile from LDS, lane m = lane%16 takes 16 consecutive K bf16
__device__ __forceinline__ v16bf loadA(const unsigned short* hbuf, int lane, int kc) {
  int m = lane & 15, half = lane >> 4;
  return *(const v16bf*)(hbuf + m * HID + kc * 32 + half * 16);
}

#define WMMA_BF16(a, b, c) \
  __builtin_amdgcn_wmma_f32_16x16x32_bf16(false, (a), false, (b), (short)0, (c), false, false)

// ---------------------------------------------------------------------------
// Persistent encoder/decoder LSTM. 64 WGs x 512 threads (16 wave32s).
// WG = 16 batch rows; wave = 16 hidden cols -> 4 gate tiles (i,f,g,o).
// h in LDS (bf16, double buffered), c in registers, weights streamed from L2.
// ---------------------------------------------------------------------------
__global__ __launch_bounds__(512)
void lstm_kernel(const float* __restrict__ x,
                 const unsigned short* __restrict__ wpack,
                 const float* __restrict__ eWih0, const float* __restrict__ eb0a, const float* __restrict__ eb0b,
                 const float* __restrict__ eb1a, const float* __restrict__ eb1b,
                 const float* __restrict__ dWih0, const float* __restrict__ db0a, const float* __restrict__ db0b,
                 const float* __restrict__ db1a, const float* __restrict__ db1b,
                 const float* __restrict__ fcW, const float* __restrict__ fcB,
                 float* __restrict__ out)
{
  __shared__ __align__(32) unsigned short lh0[2][16 * HID];
  __shared__ __align__(32) unsigned short lh1[2][16 * HID];
  __shared__ float lx[16];

  const int tid     = threadIdx.x;
  const int wave    = tid >> 5;
  const int lane    = tid & 31;
  const int col     = (wave << 4) | (lane & 15);   // hidden column 0..255
  const int rowbase = blockIdx.x << 4;             // batch-tile base row
  const int mh      = (lane >> 4) << 3;            // row offset inside C/D frag

  const unsigned short* eWhh0 = wpack + 0 * MATELEMS;
  const unsigned short* eWih1 = wpack + 1 * MATELEMS;
  const unsigned short* eWhh1 = wpack + 2 * MATELEMS;
  const unsigned short* dWhh0 = wpack + 3 * MATELEMS;
  const unsigned short* dWih1 = wpack + 4 * MATELEMS;
  const unsigned short* dWhh1 = wpack + 5 * MATELEMS;

  for (int i = tid; i < 16 * HID; i += 512) {
    lh0[0][i] = 0; lh0[1][i] = 0; lh1[0][i] = 0; lh1[1][i] = 0;
  }

  // per-lane biases and layer-0 input weights (depend only on column)
  float ebias0[4], ewx0[4], ebias1[4], dbias0[4], dwx0[4], dbias1[4];
#pragma unroll
  for (int g = 0; g < 4; ++g) {
    int n = g * HID + col;
    ebias0[g] = eb0a[n] + eb0b[n];  ewx0[g] = eWih0[n];
    ebias1[g] = eb1a[n] + eb1b[n];
    dbias0[g] = db0a[n] + db0b[n];  dwx0[g] = dWih0[n];
    dbias1[g] = db1a[n] + db1b[n];
  }

  v8f c0 = {}; v8f c1 = {};

  // ------------------------------- encoder --------------------------------
  for (int t = 0; t < TENC; ++t) {
    int r = t & 1, w = r ^ 1;
    if (tid < 16) lx[tid] = x[(rowbase + tid) * TENC + t];
    __syncthreads();

    { // layer 0: gates = h0 @ Whh0^T + x*Wih0 + bias
      v8f ai = {}, af = {}, ag = {}, ao = {};
#pragma unroll 2
      for (int kc = 0; kc < 8; ++kc) {
        v16bf a = loadA(lh0[r], lane, kc);
        ai = WMMA_BF16(a, loadB(eWhh0, lane,      wave, kc), ai);
        af = WMMA_BF16(a, loadB(eWhh0, lane, 16 + wave, kc), af);
        ag = WMMA_BF16(a, loadB(eWhh0, lane, 32 + wave, kc), ag);
        ao = WMMA_BF16(a, loadB(eWhh0, lane, 48 + wave, kc), ao);
      }
#pragma unroll
      for (int v2 = 0; v2 < 8; ++v2) {
        float xv = lx[mh + v2];
        float gi = ai[v2] + xv * ewx0[0] + ebias0[0];
        float gf = af[v2] + xv * ewx0[1] + ebias0[1];
        float gg = ag[v2] + xv * ewx0[2] + ebias0[2];
        float go = ao[v2] + xv * ewx0[3] + ebias0[3];
        float cn = sigm(gf) * c0[v2] + sigm(gi) * tanh_fast(gg);
        c0[v2] = cn;
        lh0[w][(mh + v2) * HID + col] = f2bf(sigm(go) * tanh_fast(cn));
      }
    }
    __syncthreads();

    { // layer 1: gates = h0_new @ Wih1^T + h1 @ Whh1^T + bias
      v8f ai = {}, af = {}, ag = {}, ao = {};
#pragma unroll 2
      for (int kc = 0; kc < 8; ++kc) {
        v16bf a0 = loadA(lh0[w], lane, kc);
        v16bf a1 = loadA(lh1[r], lane, kc);
        ai = WMMA_BF16(a0, loadB(eWih1, lane,      wave, kc), ai);
        ai = WMMA_BF16(a1, loadB(eWhh1, lane,      wave, kc), ai);
        af = WMMA_BF16(a0, loadB(eWih1, lane, 16 + wave, kc), af);
        af = WMMA_BF16(a1, loadB(eWhh1, lane, 16 + wave, kc), af);
        ag = WMMA_BF16(a0, loadB(eWih1, lane, 32 + wave, kc), ag);
        ag = WMMA_BF16(a1, loadB(eWhh1, lane, 32 + wave, kc), ag);
        ao = WMMA_BF16(a0, loadB(eWih1, lane, 48 + wave, kc), ao);
        ao = WMMA_BF16(a1, loadB(eWhh1, lane, 48 + wave, kc), ao);
      }
#pragma unroll
      for (int v2 = 0; v2 < 8; ++v2) {
        float gi = ai[v2] + ebias1[0];
        float gf = af[v2] + ebias1[1];
        float gg = ag[v2] + ebias1[2];
        float go = ao[v2] + ebias1[3];
        float cn = sigm(gf) * c1[v2] + sigm(gi) * tanh_fast(gg);
        c1[v2] = cn;
        lh1[w][(mh + v2) * HID + col] = f2bf(sigm(go) * tanh_fast(cn));
      }
    }
    __syncthreads();
  }

  // ------------------------------- decoder --------------------------------
  if (tid < 16) lx[tid] = x[(rowbase + tid) * TENC + (TENC - 1)];
  __syncthreads();
  // TENC is even, so decoder step d reads buffer (d & 1)

  for (int d = 0; d < TDEC; ++d) {
    int r = d & 1, w = r ^ 1;

    { // layer 0 (dec weights), input = lx (prev prediction / last x)
      v8f ai = {}, af = {}, ag = {}, ao = {};
#pragma unroll 2
      for (int kc = 0; kc < 8; ++kc) {
        v16bf a = loadA(lh0[r], lane, kc);
        ai = WMMA_BF16(a, loadB(dWhh0, lane,      wave, kc), ai);
        af = WMMA_BF16(a, loadB(dWhh0, lane, 16 + wave, kc), af);
        ag = WMMA_BF16(a, loadB(dWhh0, lane, 32 + wave, kc), ag);
        ao = WMMA_BF16(a, loadB(dWhh0, lane, 48 + wave, kc), ao);
      }
#pragma unroll
      for (int v2 = 0; v2 < 8; ++v2) {
        float xv = lx[mh + v2];
        float gi = ai[v2] + xv * dwx0[0] + dbias0[0];
        float gf = af[v2] + xv * dwx0[1] + dbias0[1];
        float gg = ag[v2] + xv * dwx0[2] + dbias0[2];
        float go = ao[v2] + xv * dwx0[3] + dbias0[3];
        float cn = sigm(gf) * c0[v2] + sigm(gi) * tanh_fast(gg);
        c0[v2] = cn;
        lh0[w][(mh + v2) * HID + col] = f2bf(sigm(go) * tanh_fast(cn));
      }
    }
    __syncthreads();

    { // layer 1 (dec weights)
      v8f ai = {}, af = {}, ag = {}, ao = {};
#pragma unroll 2
      for (int kc = 0; kc < 8; ++kc) {
        v16bf a0 = loadA(lh0[w], lane, kc);
        v16bf a1 = loadA(lh1[r], lane, kc);
        ai = WMMA_BF16(a0, loadB(dWih1, lane,      wave, kc), ai);
        ai = WMMA_BF16(a1, loadB(dWhh1, lane,      wave, kc), ai);
        af = WMMA_BF16(a0, loadB(dWih1, lane, 16 + wave, kc), af);
        af = WMMA_BF16(a1, loadB(dWhh1, lane, 16 + wave, kc), af);
        ag = WMMA_BF16(a0, loadB(dWih1, lane, 32 + wave, kc), ag);
        ag = WMMA_BF16(a1, loadB(dWhh1, lane, 32 + wave, kc), ag);
        ao = WMMA_BF16(a0, loadB(dWih1, lane, 48 + wave, kc), ao);
        ao = WMMA_BF16(a1, loadB(dWhh1, lane, 48 + wave, kc), ao);
      }
#pragma unroll
      for (int v2 = 0; v2 < 8; ++v2) {
        float gi = ai[v2] + dbias1[0];
        float gf = af[v2] + dbias1[1];
        float gg = ag[v2] + dbias1[2];
        float go = ao[v2] + dbias1[3];
        float cn = sigm(gf) * c1[v2] + sigm(gi) * tanh_fast(gg);
        c1[v2] = cn;
        lh1[w][(mh + v2) * HID + col] = f2bf(sigm(go) * tanh_fast(cn));
      }
    }
    __syncthreads();

    // fc head: pred[b] = h1[b,:] . fcW + fcb   (wave 0, split-K halves)
    if (wave == 0) {
      int m = lane & 15, half = lane >> 4;
      const unsigned short* hb = lh1[w] + m * HID + half * 128;
      const float* fw = fcW + half * 128;
      float s = 0.f;
      for (int k2 = 0; k2 < 128; ++k2) s += bf2f(hb[k2]) * fw[k2];
      s += __shfl_xor(s, 16);
      if (lane < 16) {
        float pred = s + fcB[0];
        out[(rowbase + lane) * TDEC + d] = pred;   // out is (B, HORIZON)
        lx[lane] = pred;                           // autoregressive input
      }
    }
    __syncthreads();
  }
}

extern "C" void kernel_launch(void* const* d_in, const int* in_sizes, int n_in,
                              void* d_out, int out_size, void* d_ws, size_t ws_size,
                              hipStream_t stream) {
  const float* x     = (const float*)d_in[0];
  const float* eWih0 = (const float*)d_in[1];
  const float* eWhh0 = (const float*)d_in[2];
  const float* eb0a  = (const float*)d_in[3];
  const float* eb0b  = (const float*)d_in[4];
  const float* eWih1 = (const float*)d_in[5];
  const float* eWhh1 = (const float*)d_in[6];
  const float* eb1a  = (const float*)d_in[7];
  const float* eb1b  = (const float*)d_in[8];
  const float* dWih0 = (const float*)d_in[9];
  const float* dWhh0 = (const float*)d_in[10];
  const float* db0a  = (const float*)d_in[11];
  const float* db0b  = (const float*)d_in[12];
  const float* dWih1 = (const float*)d_in[13];
  const float* dWhh1 = (const float*)d_in[14];
  const float* db1a  = (const float*)d_in[15];
  const float* db1b  = (const float*)d_in[16];
  const float* fcW   = (const float*)d_in[17];
  const float* fcB   = (const float*)d_in[18];

  unsigned short* wp = (unsigned short*)d_ws;   // needs 6 * 262144 * 2 = 3 MB

  dim3 pg(MATELEMS / 256), pb(256);
  pack_weights<<<pg, pb, 0, stream>>>(eWhh0, wp + 0 * MATELEMS);
  pack_weights<<<pg, pb, 0, stream>>>(eWih1, wp + 1 * MATELEMS);
  pack_weights<<<pg, pb, 0, stream>>>(eWhh1, wp + 2 * MATELEMS);
  pack_weights<<<pg, pb, 0, stream>>>(dWhh0, wp + 3 * MATELEMS);
  pack_weights<<<pg, pb, 0, stream>>>(dWih1, wp + 4 * MATELEMS);
  pack_weights<<<pg, pb, 0, stream>>>(dWhh1, wp + 5 * MATELEMS);

  lstm_kernel<<<64, 512, 0, stream>>>(x, wp,
                                      eWih0, eb0a, eb0b, eb1a, eb1b,
                                      dWih0, db0a, db0b, db1a, db1b,
                                      fcW, fcB, (float*)d_out);
}
